// CausalWindowAttention_52682068853011
// MI455X (gfx1250) — compile-verified, matrix operand order checked by reading, not run on hardware
//
#include <hip/hip_runtime.h>
#include <hip/hip_bf16.h>
#include <cstdint>

typedef __attribute__((ext_vector_type(16))) __bf16 v16bf;
typedef __attribute__((ext_vector_type(8)))  __bf16 v8bf;
typedef __attribute__((ext_vector_type(8)))  float  v8f;

#define B_   2
#define T_   2048
#define H_   1024
#define NH_  16
#define HD_  64
#define WIN_ 256
#define H3_  3072

// D = A(16x32 bf16) x B(32x16 bf16) + C(16x16 f32)
__device__ __forceinline__ v8f wmma_bf16(v16bf a, v16bf b, v8f c) {
  return __builtin_amdgcn_wmma_f32_16x16x32_bf16(false, a, false, b, (short)0, c,
                                                 false, false);
}

// Load a 16-bit A/B fragment for lane-half h from a contiguous K-run starting at p.
// Per ISA layout: elements 0..7 <- K = h*8+0..7 ; elements 8..15 <- K = 16+h*8+0..7
__device__ __forceinline__ v16bf load_frag(const __bf16* p, int h) {
  v8bf lo = *(const v8bf*)(p + h * 8);
  v8bf hi = *(const v8bf*)(p + 16 + h * 8);
  v16bf f;
#pragma unroll
  for (int e = 0; e < 8; ++e) { f[e] = lo[e]; f[8 + e] = hi[e]; }
  return f;
}

// ---------------- conversion kernels ----------------
__global__ void cvt_bf16_kernel(const float* __restrict__ in,
                                __bf16* __restrict__ out, int n) {
  int i = blockIdx.x * blockDim.x + threadIdx.x;
  if (i < n) out[i] = (__bf16)in[i];
}

// in[R x C] f32 -> out[C x R] bf16 (LDS-tiled). R, C multiples of 32.
__global__ void transpose_cvt(const float* __restrict__ in,
                              __bf16* __restrict__ out, int R, int C) {
  __shared__ float tile[32][33];
  int bx = blockIdx.x * 32;  // column base in `in`
  int by = blockIdx.y * 32;  // row base in `in`
  int tx = threadIdx.x;      // 0..31
  int ty = threadIdx.y;      // 0..7
#pragma unroll
  for (int yy = 0; yy < 32; yy += 8)
    tile[ty + yy][tx] = in[(size_t)(by + ty + yy) * C + (bx + tx)];
  __syncthreads();
#pragma unroll
  for (int yy = 0; yy < 32; yy += 8)
    out[(size_t)(bx + ty + yy) * R + (by + tx)] = (__bf16)tile[tx][ty + yy];
}

// ---------------- GEMM: C[MxN] = A[MxK] * Bt[NxK]^T + bias ----------------
// 8 waves / block, each wave owns a 16x64 tile.
__global__ void gemm_wmma(const __bf16* __restrict__ A, const __bf16* __restrict__ Bt,
                          const float* __restrict__ bias, void* __restrict__ Cout,
                          int M, int N, int K, int out_is_bf16) {
  const int lane = threadIdx.x & 31;
  const int wave = threadIdx.x >> 5;
  const int m = lane & 15;
  const int h = lane >> 4;
  const int row0 = blockIdx.y * 128 + wave * 16;
  const int col0 = blockIdx.x * 64;
  (void)M;
  const __bf16* arow = A + (size_t)(row0 + m) * K;
  const __bf16* brow = Bt + (size_t)(col0 + m) * K;

  v8f acc[4] = {};
  for (int k0 = 0; k0 < K; k0 += 32) {
    v16bf af = load_frag(arow + k0, h);
#pragma unroll
    for (int t = 0; t < 4; ++t) {
      v16bf bf = load_frag(brow + (size_t)(t * 16) * K + k0, h);
      acc[t] = wmma_bf16(af, bf, acc[t]);
    }
  }
#pragma unroll
  for (int t = 0; t < 4; ++t) {
    int col = col0 + t * 16 + m;
    float bv = bias[col];
#pragma unroll
    for (int r = 0; r < 8; ++r) {
      int row = row0 + r + h * 8;
      float cv = acc[t][r] + bv;
      if (out_is_bf16)
        ((__bf16*)Cout)[(size_t)row * N + col] = (__bf16)cv;
      else
        ((float*)Cout)[(size_t)row * N + col] = cv;
    }
  }
}

// ---------------- windowed causal flash attention ----------------
// qkv: bf16 [B*T, 3H] (q | k | v per row). attn: bf16 [B*T, H].
// One wave per 16-query tile; computes S^T = K*Q^T so softmax is per-lane.
__global__ void attn_wmma(const __bf16* __restrict__ qkv, __bf16* __restrict__ attn) {
  const int lane = threadIdx.x & 31;
  const int wave = threadIdx.x >> 5;
  const int m = lane & 15;   // query index in tile (== S^T column, == P row)
  const int h = lane >> 4;
  const int qt = blockIdx.x * 8 + wave;
  const int nh = blockIdx.y;
  const int b  = blockIdx.z;
  const int qbase = qt * 16;
  const int qi = qbase + m;
  const size_t bT = (size_t)b * T_;
  const int nhHD = nh * HD_;

  // Q^T fragments (B-matrix), pre-scaled by 1/sqrt(HD)=1/8 (exact in bf16)
  const __bf16* qrow = qkv + (bT + qbase + m) * (size_t)H3_ + nhHD;
  v16bf qf0 = load_frag(qrow + 0, h);
  v16bf qf1 = load_frag(qrow + 32, h);
#pragma unroll
  for (int e = 0; e < 16; ++e) {
    qf0[e] = (__bf16)((float)qf0[e] * 0.125f);
    qf1[e] = (__bf16)((float)qf1[e] * 0.125f);
  }

  float mrow = -1e30f, lrow = 0.f;
  v8f O[4] = {};

  int kstart = qbase - (WIN_ - 1);
  if (kstart < 0) kstart = 0;
  kstart &= ~31;

  const __bf16* vb = qkv + bT * (size_t)H3_ + 2 * H_ + nhHD + m;  // + j*H3 + t*16

  for (int k0 = kstart; k0 < qbase + 16; k0 += 32) {
    // --- S^T = K * Q^T, two 16-key blocks, K-dim (=HD) split 2x32 ---
    int jA0 = k0 + m;
    int jA1 = jA0 + 16;
    const __bf16* kr0 =
        qkv + (bT + (jA0 < T_ ? jA0 : T_ - 1)) * (size_t)H3_ + H_ + nhHD;
    const __bf16* kr1 =
        qkv + (bT + (jA1 < T_ ? jA1 : T_ - 1)) * (size_t)H3_ + H_ + nhHD;
    v8f s0 = {}, s1 = {};
    s0 = wmma_bf16(load_frag(kr0 + 0, h), qf0, s0);
    s0 = wmma_bf16(load_frag(kr0 + 32, h), qf1, s0);
    s1 = wmma_bf16(load_frag(kr1 + 0, h), qf0, s1);
    s1 = wmma_bf16(load_frag(kr1 + 32, h), qf1, s1);

    // --- mask + per-lane (per-query) max ---
    float tmax = -1e30f;
#pragma unroll
    for (int r = 0; r < 8; ++r) {
      int j0 = k0 + h * 8 + r;
      int j1 = j0 + 16;
      bool v0 = (j0 <= qi) && (qi - j0 < WIN_);
      bool v1 = (j1 <= qi) && (qi - j1 < WIN_);
      s0[r] = v0 ? s0[r] : -1e30f;
      s1[r] = v1 ? s1[r] : -1e30f;
      tmax = fmaxf(tmax, fmaxf(s0[r], s1[r]));
    }
    tmax = fmaxf(tmax, __shfl_xor(tmax, 16, 32));
    float newm = fmaxf(mrow, tmax);
    float alpha = __expf(mrow - newm);

    // --- probabilities; exact register match for the P A-fragment ---
    float psum = 0.f;
    v16bf pf;
#pragma unroll
    for (int r = 0; r < 8; ++r) {
      float e0 = (s0[r] > -5e29f) ? __expf(s0[r] - newm) : 0.f;
      float e1 = (s1[r] > -5e29f) ? __expf(s1[r] - newm) : 0.f;
      psum += e0 + e1;
      pf[r] = (__bf16)e0;
      pf[8 + r] = (__bf16)e1;
    }
    psum += __shfl_xor(psum, 16, 32);
    lrow = lrow * alpha + psum;
    mrow = newm;

    // broadcast alpha for O rows (row = r + h*8; lanes 0..15 hold all queries)
    float ab[8];
#pragma unroll
    for (int r = 0; r < 8; ++r) ab[r] = __shfl(alpha, r + h * 8, 32);

    // --- O = diag(alpha)*O + P * V ---
#pragma unroll
    for (int t = 0; t < 4; ++t) {
      v16bf vf;
#pragma unroll
      for (int e = 0; e < 8; ++e) {
        int j0 = k0 + h * 8 + e;
        int j1 = j0 + 16;
        if (j0 >= T_) j0 = T_ - 1;
        if (j1 >= T_) j1 = T_ - 1;
        vf[e]     = vb[(size_t)j0 * H3_ + t * 16];
        vf[8 + e] = vb[(size_t)j1 * H3_ + t * 16];
      }
      v8f o = O[t];
#pragma unroll
      for (int r = 0; r < 8; ++r) o[r] *= ab[r];
      O[t] = wmma_bf16(pf, vf, o);
    }
  }

  float linv = 1.f / lrow;
  float lb[8];
#pragma unroll
  for (int r = 0; r < 8; ++r) lb[r] = __shfl(linv, r + h * 8, 32);
#pragma unroll
  for (int t = 0; t < 4; ++t) {
#pragma unroll
    for (int r = 0; r < 8; ++r) {
      int row = qbase + r + h * 8;
      attn[(bT + row) * (size_t)H_ + nhHD + t * 16 + m] = (__bf16)(O[t][r] * lb[r]);
    }
  }
}

extern "C" void kernel_launch(void* const* d_in, const int* in_sizes, int n_in,
                              void* d_out, int out_size, void* d_ws, size_t ws_size,
                              hipStream_t stream) {
  (void)in_sizes; (void)n_in; (void)out_size; (void)ws_size;
  const float* x    = (const float*)d_in[0];
  const float* Wqkv = (const float*)d_in[1];
  const float* bqkv = (const float*)d_in[2];
  const float* Wout = (const float*)d_in[3];
  const float* bout = (const float*)d_in[4];

  char* ws = (char*)d_ws;
  size_t off = 0;
  __bf16* x_bf    = (__bf16*)(ws + off); off += (size_t)B_ * T_ * H_ * 2;   // 8 MB
  __bf16* wqkv_t  = (__bf16*)(ws + off); off += (size_t)H_ * H3_ * 2;       // 6 MB
  __bf16* wout_t  = (__bf16*)(ws + off); off += (size_t)H_ * H_ * 2;        // 2 MB
  __bf16* qkv_bf  = (__bf16*)(ws + off); off += (size_t)B_ * T_ * H3_ * 2;  // 24 MB
  __bf16* attn_bf = (__bf16*)(ws + off); off += (size_t)B_ * T_ * H_ * 2;   // 8 MB

  const int nx = B_ * T_ * H_;
  cvt_bf16_kernel<<<(nx + 255) / 256, 256, 0, stream>>>(x, x_bf, nx);
  transpose_cvt<<<dim3(H3_ / 32, H_ / 32), dim3(32, 8), 0, stream>>>(Wqkv, wqkv_t, H_, H3_);
  transpose_cvt<<<dim3(H_ / 32, H_ / 32), dim3(32, 8), 0, stream>>>(Wout, wout_t, H_, H_);

  // qkv = x @ W_qkv + b  -> bf16 [B*T, 3H]
  gemm_wmma<<<dim3(H3_ / 64, (B_ * T_) / 128), 256, 0, stream>>>(
      x_bf, wqkv_t, bqkv, qkv_bf, B_ * T_, H3_, H_, 1);

  // windowed causal attention -> bf16 [B*T, H]
  attn_wmma<<<dim3(T_ / 128, NH_, B_), 256, 0, stream>>>(qkv_bf, attn_bf);

  // out = attn @ W_out + b -> f32 [B*T, H]
  gemm_wmma<<<dim3(H_ / 64, (B_ * T_) / 128), 256, 0, stream>>>(
      attn_bf, wout_t, bout, (float*)d_out, B_ * T_, H_, H_, 0);
}
